// AttendCompareAggregate_14422500180160
// MI455X (gfx1250) — matrix operand order repaired and checked by reading, not compiled, not run on hardware
//
#include <hip/hip_runtime.h>
#include <hip/hip_bf16.h>

typedef __attribute__((ext_vector_type(16))) __bf16 v16bf;
typedef __attribute__((ext_vector_type(8)))  float  v8f;

union Frag32B { v16bf v; uint4 q[2]; };

__device__ __forceinline__ unsigned short f32_to_bf16(float f) {
    union { float f; unsigned int u; } c; c.f = f;
    unsigned int u = c.u;
    unsigned int r = u + 0x7FFFu + ((u >> 16) & 1u);
    if ((u & 0x7F800000u) == 0x7F800000u) r = u;   // Inf/NaN: truncate
    return (unsigned short)(r >> 16);
}

// ---------------------------------------------------------------------------
// Layout converters (f32 -> bf16)
// ---------------------------------------------------------------------------
// q/p: (L, B, D) f32 -> (B, L, D) bf16   (L=512, B=64, D=512)
__global__ __launch_bounds__(256) void conv_qp(const float* __restrict__ src,
                                               unsigned short* __restrict__ dst) {
    long long idx = (long long)blockIdx.x * 256 + threadIdx.x;  // over L*B*D
    int d = (int)(idx & 511);
    long long t = idx >> 9;
    int b = (int)(t & 63);
    long long l = t >> 6;
    dst[((long long)b * 512 + l) * 512 + d] = f32_to_bf16(src[idx]);
}

__global__ __launch_bounds__(256) void conv_elem(const float* __restrict__ src,
                                                 unsigned short* __restrict__ dst,
                                                 long long n) {
    long long i = (long long)blockIdx.x * 256 + threadIdx.x;
    if (i < n) dst[i] = f32_to_bf16(src[i]);
}

// ---------------------------------------------------------------------------
// NT GEMM: C(M x N) = A(M x K) * Bt(N x K)^T, bf16 inputs, f32 accumulate.
// K fixed = 512. ldA = ldB = 512, ldC = N. Optional batched via blockIdx.z.
// Block tile 128x128, 8 waves (wave32), each wave 64x32 (4x2 WMMA frags).
// Staging: GLOBAL_LOAD_ASYNC_TO_LDS_B128 (ASYNCcnt) + double-buffered LDS.
// ---------------------------------------------------------------------------
#define BM 128
#define BN 128
#define BK 32
#define LDT 40   // padded LDS row stride (elements) -> 80B, 16B aligned

template <bool OUT_BF16>
__global__ __launch_bounds__(256) void gemm_nt_bf16(
    const unsigned short* __restrict__ A,
    const unsigned short* __restrict__ Bt,
    float* __restrict__ Cf,
    unsigned short* __restrict__ Cb,
    const float* __restrict__ bias,
    int N, long long strideA, long long strideB, long long strideC)
{
    __shared__ unsigned short sA[2][BM * LDT];
    __shared__ unsigned short sB[2][BN * LDT];

    const int K    = 512;
    const int tid  = threadIdx.x;
    const int lane = tid & 31;
    const int wave = tid >> 5;
    const int wm   = wave & 1;    // 0..1 -> 64-row slab
    const int wn   = wave >> 1;   // 0..3 -> 32-col slab
    const int hl   = lane >> 4;   // lane half
    const int l16  = lane & 15;
    const int z    = blockIdx.z;

    const unsigned short* Ab = A  + z * strideA + (long long)blockIdx.y * BM * K;
    const unsigned short* Bb = Bt + z * strideB + (long long)blockIdx.x * BN * K;
    const unsigned long long Abase = (unsigned long long)Ab;
    const unsigned long long Bbase = (unsigned long long)Bb;

    // Per-thread tile chunks: 512 chunks of 8 bf16 (16B), 2 per thread.
    const int r0  = tid >> 2;                 // 0..63
    const int r1  = r0 + 64;                  // 64..127
    const int cc0 = (tid & 3) << 3;           // 0,8,16,24

    // Issue async DMA of one 128x32 bf16 tile pair into LDS buffer `buf`.
    auto stage = [&](int buf, int kt) {
        unsigned offA0 = (unsigned)((r0 * 512 + kt + cc0) * 2);
        unsigned offA1 = (unsigned)((r1 * 512 + kt + cc0) * 2);
        unsigned la0 = (unsigned)(size_t)&sA[buf][r0 * LDT + cc0];
        unsigned la1 = (unsigned)(size_t)&sA[buf][r1 * LDT + cc0];
        unsigned lb0 = (unsigned)(size_t)&sB[buf][r0 * LDT + cc0];
        unsigned lb1 = (unsigned)(size_t)&sB[buf][r1 * LDT + cc0];
        asm volatile("global_load_async_to_lds_b128 %0, %1, %2"
                     :: "v"(la0), "v"(offA0), "s"(Abase) : "memory");
        asm volatile("global_load_async_to_lds_b128 %0, %1, %2"
                     :: "v"(la1), "v"(offA1), "s"(Abase) : "memory");
        asm volatile("global_load_async_to_lds_b128 %0, %1, %2"
                     :: "v"(lb0), "v"(offA0), "s"(Bbase) : "memory");
        asm volatile("global_load_async_to_lds_b128 %0, %1, %2"
                     :: "v"(lb1), "v"(offA1), "s"(Bbase) : "memory");
    };

    v8f acc[4][2];
#pragma unroll
    for (int i = 0; i < 4; ++i)
#pragma unroll
        for (int j = 0; j < 2; ++j)
#pragma unroll
            for (int v = 0; v < 8; ++v) acc[i][j][v] = 0.0f;

    stage(0, 0);

    int buf = 0;
    for (int kt = 0; kt < K; kt += BK, buf ^= 1) {
        // Wait for this wave's DMA into `buf`, then rendezvous: after the
        // barrier every wave's tile data is resident in LDS.
        asm volatile("s_wait_asynccnt 0x0" ::: "memory");
        __syncthreads();

        // Kick off DMA of the next tile into the other buffer; it overlaps
        // with the WMMA work below (disjoint LDS buffer, own ASYNCcnt).
        if (kt + BK < K) stage(buf ^ 1, kt + BK);

        // B fragments: lane l16 = column n, VGPR j holds B[k0+2j..2j+1][n],
        // k0 = hl*16 -> 16 contiguous K elems per lane from the N x K layout.
        Frag32B bfr[2];
#pragma unroll
        for (int j = 0; j < 2; ++j) {
            int n = wn * 32 + j * 16 + l16;
            const uint4* pb = (const uint4*)&sB[buf][n * LDT + hl * 16];
            bfr[j].q[0] = pb[0];
            bfr[j].q[1] = pb[1];
        }

#pragma unroll
        for (int i = 0; i < 4; ++i) {
            // A fragment: row m, VGPR 0-3 = K[hl*16 .. +7], VGPR 4-7 = K[hl*16+8 .. +7]
            int m = wm * 64 + i * 16 + l16;
            Frag32B af;
            af.q[0] = *(const uint4*)&sA[buf][m * LDT + hl * 16];
            af.q[1] = *(const uint4*)&sA[buf][m * LDT + hl * 16 + 8];
#pragma unroll
            for (int j = 0; j < 2; ++j)
                acc[i][j] = __builtin_amdgcn_wmma_f32_16x16x32_bf16(
                    false, af.v, false, bfr[j].v, (short)0, acc[i][j], false, false);
        }
    }

    // Epilogue. C layout: VGPR v -> row (v + 8*hl), col = l16.
#pragma unroll
    for (int i = 0; i < 4; ++i) {
#pragma unroll
        for (int j = 0; j < 2; ++j) {
            long long col = (long long)blockIdx.x * BN + wn * 32 + j * 16 + l16;
            float bv = bias ? bias[col] : 0.0f;
#pragma unroll
            for (int v = 0; v < 8; ++v) {
                long long row = (long long)blockIdx.y * BM + wm * 64 + i * 16 + hl * 8 + v;
                long long idx = z * strideC + row * (long long)N + col;
                float val = acc[i][j][v] + bv;
                if (OUT_BF16) Cb[idx] = f32_to_bf16(val);
                else          Cf[idx] = val;
            }
        }
    }
}

// ---------------------------------------------------------------------------
// Softmax statistics and marginal sums over the score tensor e (B, 512, 512)
// ---------------------------------------------------------------------------
__global__ __launch_bounds__(256) void row_stats(const float* __restrict__ e,
                                                 float* __restrict__ rowM,
                                                 float* __restrict__ rowZ) {
    __shared__ float sb[256];
    const float* r = e + (long long)blockIdx.x * 512;
    int t = threadIdx.x;
    float x0 = r[t], x1 = r[t + 256];
    float m = fmaxf(x0, x1);
    sb[t] = m; __syncthreads();
    for (int s = 128; s > 0; s >>= 1) { if (t < s) sb[t] = fmaxf(sb[t], sb[t + s]); __syncthreads(); }
    m = sb[0]; __syncthreads();
    float zv = __expf(x0 - m) + __expf(x1 - m);
    sb[t] = zv; __syncthreads();
    for (int s = 128; s > 0; s >>= 1) { if (t < s) sb[t] += sb[t + s]; __syncthreads(); }
    if (t == 0) { rowM[blockIdx.x] = m; rowZ[blockIdx.x] = sb[0]; }
}

// Fused single pass per column p: online softmax stats over q (colM/colZ)
// AND cs[b,p] = sum_q softmax_over_p(e[b,q,:])[p]  (uses rowM/rowZ).
__global__ __launch_bounds__(512) void colstats_cs(const float* __restrict__ e,
                                                   const float* __restrict__ rowM,
                                                   const float* __restrict__ rowZ,
                                                   float* __restrict__ colM,
                                                   float* __restrict__ colZ,
                                                   float* __restrict__ cs) {
    int b = blockIdx.x, pcol = threadIdx.x;
    const float* eb = e + (long long)b * 262144;
    const float* rm = rowM + b * 512;
    const float* rz = rowZ + b * 512;
    float m = -3.0e38f, zv = 0.0f, acc = 0.0f;
    for (int qi = 0; qi < 512; ++qi) {
        float x = eb[qi * 512 + pcol];
        float nm = fmaxf(m, x);
        zv = zv * __expf(m - nm) + __expf(x - nm);
        m = nm;
        acc += __expf(x - rm[qi]) / rz[qi];
    }
    colM[b * 512 + pcol] = m;
    colZ[b * 512 + pcol] = zv;
    cs[b * 512 + pcol] = acc;
}

// rs[b,q] = sum_p softmax_over_q(e[b,:,p])[q]
__global__ __launch_bounds__(256) void rs_kernel(const float* __restrict__ e,
                                                 const float* __restrict__ colM,
                                                 const float* __restrict__ colZ,
                                                 float* __restrict__ rs) {
    __shared__ float sb[256];
    int bq = blockIdx.x;              // b*512 + q
    int b = bq >> 9;
    const float* r  = e + (long long)bq * 512;
    const float* cm = colM + b * 512;
    const float* cz = colZ + b * 512;
    int t = threadIdx.x;
    float a = __expf(r[t] - cm[t]) / cz[t]
            + __expf(r[t + 256] - cm[t + 256]) / cz[t + 256];
    sb[t] = a; __syncthreads();
    for (int s = 128; s > 0; s >>= 1) { if (t < s) sb[t] += sb[t + s]; __syncthreads(); }
    if (t == 0) rs[bq] = sb[0];
}

// u1 = WG^T @ WH[0,:512], u2 = WG^T @ WH[0,512:]   (each length 1024)
__global__ __launch_bounds__(256) void u12_kernel(const float* __restrict__ WG,
                                                  const float* __restrict__ WH,
                                                  float* __restrict__ u1,
                                                  float* __restrict__ u2) {
    int j = blockIdx.x * 256 + threadIdx.x;   // 0..1023
    float a1 = 0.0f, a2 = 0.0f;
    for (int d = 0; d < 512; ++d) {
        float w = WG[d * 1024 + j];
        a1 += WH[d]       * w;
        a2 += WH[512 + d] * w;
    }
    u1[j] = a1; u2[j] = a2;
}

// out[b] = sumq.u1lo + <cs, pb.u1hi> + sump.u2lo + <rs, qb.u2hi> + c0
__global__ __launch_bounds__(512) void final_kernel(
    const float* __restrict__ q, const float* __restrict__ p,
    const float* __restrict__ cs, const float* __restrict__ rs,
    const float* __restrict__ u1, const float* __restrict__ u2,
    const float* __restrict__ bG, const float* __restrict__ WH,
    const float* __restrict__ bH, float* __restrict__ out)
{
    __shared__ float sb[512];
    __shared__ float su1h[512];
    __shared__ float su2h[512];
    int b = blockIdx.x, t = threadIdx.x;
    su1h[t] = u1[512 + t];
    su2h[t] = u2[512 + t];
    __syncthreads();

    // feature sums over positions (coalesced in t = d)
    float sq = 0.0f, sp = 0.0f;
    for (int l = 0; l < 512; ++l) {
        sq += q[((long long)l * 64 + b) * 512 + t];
        sp += p[((long long)l * 64 + b) * 512 + t];
    }
    float part = sq * u1[t] + sp * u2[t];

    const float* prow = p + ((long long)t * 64 + b) * 512;  // pb[b, t, :]
    float tp = 0.0f;
    for (int d = 0; d < 512; ++d) tp += prow[d] * su1h[d];
    part += cs[b * 512 + t] * tp;

    const float* qrow = q + ((long long)t * 64 + b) * 512;  // qb[b, t, :]
    float tq = 0.0f;
    for (int d = 0; d < 512; ++d) tq += qrow[d] * su2h[d];
    part += rs[b * 512 + t] * tq;

    sb[t] = part; __syncthreads();
    for (int s = 256; s > 0; s >>= 1) { if (t < s) sb[t] += sb[t + s]; __syncthreads(); }
    if (t == 0) {
        float c0 = 0.0f;
        for (int d = 0; d < 512; ++d) c0 += bG[d] * (WH[d] + WH[512 + d]);
        out[b] = sb[0] + 512.0f * c0 + bH[0];
    }
}

// ---------------------------------------------------------------------------
extern "C" void kernel_launch(void* const* d_in, const int* in_sizes, int n_in,
                              void* d_out, int out_size, void* d_ws, size_t ws_size,
                              hipStream_t stream) {
    const float* q  = (const float*)d_in[0];
    const float* p  = (const float*)d_in[1];
    const float* WF = (const float*)d_in[2];
    const float* bF = (const float*)d_in[3];
    const float* WG = (const float*)d_in[4];
    const float* bG = (const float*)d_in[5];
    const float* WH = (const float*)d_in[6];
    const float* bH = (const float*)d_in[7];
    float* out = (float*)d_out;

    const long long NB = 64, L = 512, D = 512;
    const long long QPD = NB * L * D;   // 16,777,216 elements

    char* ws = (char*)d_ws;
    unsigned short* qbf  = (unsigned short*)ws; ws += QPD * 2;
    unsigned short* pbf  = (unsigned short*)ws; ws += QPD * 2;
    unsigned short* wfbf = (unsigned short*)ws; ws += D * D * 2;
    unsigned short* attq = (unsigned short*)ws; ws += QPD * 2;
    unsigned short* attp = (unsigned short*)ws; ws += QPD * 2;
    float* e    = (float*)ws; ws += NB * L * L * 4;
    float* rowM = (float*)ws; ws += NB * L * 4;
    float* rowZ = (float*)ws; ws += NB * L * 4;
    float* colM = (float*)ws; ws += NB * L * 4;
    float* colZ = (float*)ws; ws += NB * L * 4;
    float* cs   = (float*)ws; ws += NB * L * 4;
    float* rs   = (float*)ws; ws += NB * L * 4;
    float* u1   = (float*)ws; ws += 1024 * 4;
    float* u2   = (float*)ws; ws += 1024 * 4;

    // 1) convert q, p to (B, L, D) bf16; WF to bf16
    conv_qp<<<(int)(QPD / 256), 256, 0, stream>>>(q, qbf);
    conv_qp<<<(int)(QPD / 256), 256, 0, stream>>>(p, pbf);
    conv_elem<<<(int)((D * D + 255) / 256), 256, 0, stream>>>(WF, wfbf, D * D);

    // 2) att_q / att_p = X @ WF^T + bF  (NT GEMM, M = B*L = 32768, N = 512)
    dim3 gAtt(512 / BN, (unsigned)((NB * L) / BM), 1);
    gemm_nt_bf16<true><<<gAtt, 256, 0, stream>>>(qbf, wfbf, nullptr, attq, bF,
                                                 512, 0, 0, 0);
    gemm_nt_bf16<true><<<gAtt, 256, 0, stream>>>(pbf, wfbf, nullptr, attp, bF,
                                                 512, 0, 0, 0);

    // 3) e[b] = att_q[b] @ att_p[b]^T  (batched NT GEMM, f32 out)
    dim3 gE(512 / BN, 512 / BM, (unsigned)NB);
    gemm_nt_bf16<false><<<gE, 256, 0, stream>>>(attq, attp, e, nullptr, nullptr,
                                                512, L * D, L * D, L * L);

    // 4) softmax stats both directions + marginal sums (3 passes over e)
    row_stats<<<(int)(NB * L), 256, 0, stream>>>(e, rowM, rowZ);
    colstats_cs<<<(int)NB, 512, 0, stream>>>(e, rowM, rowZ, colM, colZ, cs);
    rs_kernel<<<(int)(NB * L), 256, 0, stream>>>(e, colM, colZ, rs);

    // 5) collapsed Compare+Aggregate via u1/u2 and dot products
    u12_kernel<<<4, 256, 0, stream>>>(WG, WH, u1, u2);
    final_kernel<<<(int)NB, 512, 0, stream>>>(q, p, cs, rs, u1, u2, bG, WH, bH, out);
}
